// PostProcessCocoGrounding_7404523618811
// MI455X (gfx1250) — matrix operand order repaired and checked by reading, not compiled, not run on hardware
//
#include <hip/hip_runtime.h>
#include <hip/hip_bf16.h>

typedef __attribute__((ext_vector_type(16))) _Float16 v16h;
typedef __attribute__((ext_vector_type(8)))  float    v8f;

// Problem constants (from reference)
constexpr int Bn   = 128;
constexpr int NQ   = 900;
constexpr int Tt   = 256;
constexpr int Cc   = 1203;
constexpr int NSEL = 300;

constexpr int MT     = (NQ + 15) / 16;   // 57 q-tiles (only last is partial)
constexpr int NT     = (Cc + 15) / 16;   // 76 c-tiles (only last is partial)
constexpr int NTILES = MT * NT;          // 4332
constexpr int KS     = Tt / 32;          // 8 K-steps
constexpr int NBINS  = 1024;
constexpr int CANDMX = 3072;
constexpr int NTHR   = 256;              // 8 waves (wave32)
constexpr int NWAVES = NTHR / 32;

// Single v_rcp_f32 + v_exp_f32: feeds f16 WMMA operands, 1-ulp rcp is plenty.
__device__ __forceinline__ float fast_sigmoid(float x) {
    return __builtin_amdgcn_rcpf(1.0f + __expf(-x));
}

// One 16x16 output tile: 8 chained WMMAs over K=256 with A cached in VGPRs.
// BMASK=true only for the single partial c-tile; elsewhere no mask VALU work.
template <bool BMASK>
__device__ __forceinline__ v8f gemm_tile(const v16h (&Acache)[KS],
                                         const float* __restrict__ pb_row,
                                         float bmsk) {
    v8f acc = {};
#pragma unroll
    for (int ks = 0; ks < KS; ++ks) {
        const float* pb = pb_row + ks * 32;
        v16h bm;
#pragma unroll
        for (int j = 0; j < 16; ++j) {
            float x = pb[j + (j & 8)];
            if (BMASK) x *= bmsk;
            bm[j] = (_Float16)x;
        }
        acc = __builtin_amdgcn_wmma_f32_16x16x32_f16(
                false, Acache[ks], false, bm, (short)0, acc, false, false);
    }
    return acc;
}

__global__ void __launch_bounds__(NTHR)
postprocess_grounding_kernel(const float* __restrict__ logits,   // [B,NQ,T]
                             const float* __restrict__ bbox,     // [B,NQ,4]
                             const float* __restrict__ pos_maps, // [C,T]
                             const float* __restrict__ tsizes,   // [B,2] (h,w)
                             const int*   __restrict__ nselp,
                             float* __restrict__ out)
{
    __shared__ unsigned int hist[NBINS];
    __shared__ float        tile_max[NTILES];
    __shared__ float        cand_val[CANDMX];
    __shared__ int          cand_idx[CANDMX];
    __shared__ unsigned int cand_count;
    __shared__ int          thresh_bin;
    __shared__ float        red_val[NTHR];
    __shared__ int          red_flat[NTHR];
    __shared__ int          red_slot[NTHR];
    __shared__ float        top_val[NSEL];
    __shared__ int          top_idx[NSEL];

    const int b      = blockIdx.x;
    const int tid    = threadIdx.x;
    const int wave   = tid >> 5;
    const int lane   = tid & 31;
    const int lane_m = lane & 15;  // m (A row) or n (B/C col)
    const int lane_h = lane >> 4;  // half selector

    const int ns = min(*nselp, NSEL);
    const float* __restrict__ lg = logits + (size_t)b * NQ * Tt;

    for (int i = tid; i < NBINS; i += NTHR) hist[i] = 0u;
    if (tid == 0) cand_count = 0u;
    __syncthreads();

    // ---- two passes: 0 = histogram + tile-max, 1 = candidate collection ----
    for (int phase = 0; phase < 2; ++phase) {
        const float th_lo = (phase == 1)
                          ? (float)thresh_bin * (1.0f / (float)NBINS)
                          : 0.0f;

        // Each wave owns whole m-tiles: A (16x256) is sigmoid-converted ONCE
        // into 64 VGPRs and reused across all 76 c-tiles.
        for (int mt = wave; mt < MT; mt += NWAVES) {
            const int q0 = mt * 16;

            if (phase == 1) {
                // Skip the whole m-row if no tile in it can reach threshold.
                bool any = false;
                for (int nt = 0; nt < NT; ++nt)
                    any = any || (tile_max[mt * NT + nt] >= th_lo);
                if (!any) continue;
            }

            // ---- load + sigmoid A once per m-tile ----
            const int   qa   = q0 + lane_m;
            const float amsk = (qa < NQ) ? 1.0f : 0.0f;
            const float* __restrict__ pa_row =
                lg + (size_t)min(qa, NQ - 1) * Tt + lane_h * 8;

            v16h Acache[KS];
            if (mt != MT - 1) {
#pragma unroll
                for (int ks = 0; ks < KS; ++ks) {
                    const float* pa = pa_row + ks * 32;
                    __builtin_prefetch(pa + 32, 0, 1);
#pragma unroll
                    for (int j = 0; j < 8; ++j) {
                        Acache[ks][j]     = (_Float16)fast_sigmoid(pa[j]);
                        Acache[ks][j + 8] = (_Float16)fast_sigmoid(pa[16 + j]);
                    }
                }
            } else {
#pragma unroll
                for (int ks = 0; ks < KS; ++ks) {
                    const float* pa = pa_row + ks * 32;
#pragma unroll
                    for (int j = 0; j < 8; ++j) {
                        Acache[ks][j]     = (_Float16)(fast_sigmoid(pa[j])      * amsk);
                        Acache[ks][j + 8] = (_Float16)(fast_sigmoid(pa[16 + j]) * amsk);
                    }
                }
            }

            // ---- sweep all c-tiles against cached A ----
            for (int nt = 0; nt < NT; ++nt) {
                const int t = mt * NT + nt;
                if (phase == 1 && tile_max[t] < th_lo) continue;  // uniform

                const int   c0   = nt * 16;
                const int   cb   = c0 + lane_m;
                const float bmsk = (cb < Cc) ? 1.0f : 0.0f;
                const float* __restrict__ pb_row =
                    pos_maps + (size_t)min(cb, Cc - 1) * Tt + lane_h * 8;

                // Only the final c-tile is partial: keep the hot path mask-free.
                const v8f acc = (nt != NT - 1)
                              ? gemm_tile<false>(Acache, pb_row, bmsk)
                              : gemm_tile<true>(Acache, pb_row, bmsk);

                // consume 16x16 result: lane = col n, rows r + 8*lane_h
                const int c = c0 + lane_m;
                if (phase == 0) {
                    float wmax = -1e30f;
#pragma unroll
                    for (int r = 0; r < 8; ++r) {
                        const int q = q0 + r + 8 * lane_h;
                        const float v = acc[r];
                        wmax = fmaxf(wmax, v);
                        if (q < NQ && c < Cc) {
                            const float vc = fminf(fmaxf(v, 0.0f), 0.99999988f);
                            atomicAdd(&hist[(int)(vc * (float)NBINS)], 1u);
                        }
                    }
                    // cross-lane max (wave32) -> tile_max
#pragma unroll
                    for (int off = 16; off > 0; off >>= 1)
                        wmax = fmaxf(wmax, __shfl_xor(wmax, off, 32));
                    if (lane == 0) tile_max[t] = wmax;
                } else {
#pragma unroll
                    for (int r = 0; r < 8; ++r) {
                        const int q = q0 + r + 8 * lane_h;
                        const float v  = acc[r];
                        const float vc = fminf(fmaxf(v, 0.0f), 0.99999988f);
                        if (q < NQ && c < Cc &&
                            (int)(vc * (float)NBINS) >= thresh_bin) {
                            const unsigned int slot = atomicAdd(&cand_count, 1u);
                            if (slot < (unsigned)CANDMX) {
                                cand_val[slot] = v;
                                cand_idx[slot] = q * Cc + c;
                            }
                        }
                    }
                }
            }
        }
        __syncthreads();
        if (phase == 0) {
            if (tid == 0) {
                unsigned int cum = 0; int th = 0;
                for (int i = NBINS - 1; i >= 0; --i) {
                    cum += hist[i];
                    if (cum >= (unsigned)ns) { th = i; break; }
                }
                thresh_bin = th;
            }
            __syncthreads();
        }
    }

    // ---- exact top-ns selection over candidates (deterministic tie-break) ----
    const int ncand = min((int)cand_count, CANDMX);
    for (int s = 0; s < ns; ++s) {
        float bestv = -1e30f; int bestf = 0x7fffffff; int bests = -1;
        for (int i = tid; i < ncand; i += NTHR) {
            const float v = cand_val[i];
            const int   f = cand_idx[i];
            if (v > bestv || (v == bestv && f < bestf)) {
                bestv = v; bestf = f; bests = i;
            }
        }
        red_val[tid] = bestv; red_flat[tid] = bestf; red_slot[tid] = bests;
        __syncthreads();
        for (int off = NTHR / 2; off > 0; off >>= 1) {
            if (tid < off) {
                const float v2 = red_val[tid + off];
                const int   f2 = red_flat[tid + off];
                if (v2 > red_val[tid] ||
                    (v2 == red_val[tid] && f2 < red_flat[tid])) {
                    red_val[tid]  = v2;
                    red_flat[tid] = f2;
                    red_slot[tid] = red_slot[tid + off];
                }
            }
            __syncthreads();
        }
        if (tid == 0) {
            const int slot = red_slot[0];
            if (slot >= 0) {
                top_val[s] = red_val[0];
                top_idx[s] = red_flat[0];
                cand_val[slot] = -2.0f;   // consume
            } else {
                top_val[s] = 0.0f;
                top_idx[s] = 0;
            }
        }
        __syncthreads();
    }

    // ---- write scores / labels / boxes ----
    const float img_h = tsizes[b * 2 + 0];
    const float img_w = tsizes[b * 2 + 1];
    float* __restrict__ out_scores = out;
    float* __restrict__ out_labels = out + (size_t)Bn * ns;
    float* __restrict__ out_boxes  = out + 2 * (size_t)Bn * ns;

    for (int i = tid; i < ns; i += NTHR) {
        const float sc = top_val[i];
        const int   fi = top_idx[i];
        const int   q  = fi / Cc;
        const int   cl = fi % Cc;
        out_scores[(size_t)b * ns + i] = sc;
        out_labels[(size_t)b * ns + i] = (float)cl;
        const float* bb = bbox + ((size_t)b * NQ + q) * 4;
        const float cx = bb[0], cy = bb[1], w = bb[2], h = bb[3];
        float* ob = out_boxes + ((size_t)b * ns + i) * 4;
        ob[0] = (cx - 0.5f * w) * img_w;
        ob[1] = (cy - 0.5f * h) * img_h;
        ob[2] = (cx + 0.5f * w) * img_w;
        ob[3] = (cy + 0.5f * h) * img_h;
    }
}

extern "C" void kernel_launch(void* const* d_in, const int* in_sizes, int n_in,
                              void* d_out, int out_size, void* d_ws, size_t ws_size,
                              hipStream_t stream) {
    const float* logits   = (const float*)d_in[0];
    const float* bbox     = (const float*)d_in[1];
    const float* pos_maps = (const float*)d_in[2];
    const float* tsizes   = (const float*)d_in[3];
    const int*   nsel     = (const int*)d_in[4];
    float* out = (float*)d_out;
    (void)in_sizes; (void)n_in; (void)out_size; (void)d_ws; (void)ws_size;

    postprocess_grounding_kernel<<<Bn, NTHR, 0, stream>>>(
        logits, bbox, pos_maps, tsizes, nsel, out);
}
